// LiquidStateMachine_62775241999147
// MI455X (gfx1250) — matrix operand order repaired and checked by reading, not compiled, not run on hardware
//
#include <hip/hip_runtime.h>
#include <hip/hip_bf16.h>

typedef __attribute__((ext_vector_type(16))) _Float16 v16h;
typedef __attribute__((ext_vector_type(8)))  _Float16 v8h;
typedef __attribute__((ext_vector_type(8)))  float    v8f;

union H16 { v16h v; v8h h[2]; };

// Problem constants (from reference)
#define LSM_B 64
#define LSM_T 256
#define LSM_D 512
#define LSM_R 2048
#define LSM_K (LSM_D + LSM_R)   // 2560
#define ALPHA_F 0.9f

// ---------------------------------------------------------------------------
// One-time prep: convert inputs [B,T,D] f32 -> f16
__global__ __launch_bounds__(256) void lsm_convert_x(const float* __restrict__ x,
                                                     _Float16* __restrict__ xh,
                                                     int n) {
    int i = blockIdx.x * 256 + threadIdx.x;
    if (i < n) xh[i] = (_Float16)x[i];
}

// One-time prep: build Wt [R][K] f16 row-major, where
//   Wt[n][k]       = W_in[k][n]        for k < 512
//   Wt[n][512 + j] = W_rec[j][n]       for j < 2048
// i.e. row n of Wt is the n-th column of [W_in ; W_rec]. This makes the WMMA
// B-fragment (lane = output column, 16 contiguous K halves) a contiguous load.
__global__ __launch_bounds__(256) void lsm_build_wt(const float* __restrict__ W_in,
                                                    const float* __restrict__ W_rec,
                                                    _Float16* __restrict__ Wt) {
    int i = blockIdx.x * 256 + threadIdx.x;      // 0 .. R*K-1
    int n = i & (LSM_R - 1);                     // coalesced reads across n
    int k = i >> 11;                             // 0 .. 2559
    float v = (k < LSM_D) ? W_in[(size_t)k * LSM_R + n]
                          : W_rec[(size_t)(k - LSM_D) * LSM_R + n];
    Wt[(size_t)n * LSM_K + k] = (_Float16)v;
}

// Per-call state reset (u, spike count, both spike buffers)
__global__ __launch_bounds__(256) void lsm_init(float* __restrict__ u,
                                                float* __restrict__ cnt,
                                                _Float16* __restrict__ s0,
                                                _Float16* __restrict__ s1,
                                                int n) {
    int i = blockIdx.x * 256 + threadIdx.x;
    if (i < n) { u[i] = 0.0f; cnt[i] = 0.0f; s0[i] = (_Float16)0.0f; s1[i] = (_Float16)0.0f; }
}

// ---------------------------------------------------------------------------
// One timestep:  I = x_t @ W_in + s_in @ W_rec  (f16 WMMA, f32 accum),
// then LIF update:  u = ALPHA*u + I; spike = (u >= 1); u *= (1-spike).
// Grid: 64 blocks x 256 threads = 512 wave32s; wave gw owns C tile
// (m_tile = gw&3, n_tile = gw>>2) of the [64][2048] state.
__global__ __launch_bounds__(256) void lsm_step(const _Float16* __restrict__ Xh,  // [B,T,D]
                                                const _Float16* __restrict__ Wt,  // [R][K]
                                                const _Float16* __restrict__ Sin, // [B][R]
                                                _Float16* __restrict__ Sout,      // [B][R]
                                                float* __restrict__ u,            // [B][R]
                                                float* __restrict__ cnt,          // [B][R]
                                                int t) {
    const int lane = threadIdx.x & 31;
    const int wave = threadIdx.x >> 5;
    const int gw   = blockIdx.x * 8 + wave;   // 0..511
    const int m0   = (gw & 3) * 16;           // M tile base (0..48)
    const int n0   = (gw >> 2) * 16;          // N tile base (0..2032)
    const int lrow = lane & 15;
    const int lhi  = lane >> 4;

    v8f c = {};

    // A fragment source rows (row-major, exactly-representable {0,1} halves)
    const _Float16* arow = Xh + ((size_t)(m0 + lrow) * LSM_T + t) * LSM_D; // x_t row
    const _Float16* srow = Sin + (size_t)(m0 + lrow) * LSM_R;             // spike row
    // B fragment source row: Wt row n holds column n of [W_in ; W_rec]
    const _Float16* brow = Wt + (size_t)(n0 + lrow) * LSM_K;

    // Phase 1: x_t @ W_in, K = 512
    for (int kb = 0; kb < LSM_D; kb += 32) {
        H16 a, b;
        a.h[0] = *(const v8h*)(arow + kb + lhi * 16);
        a.h[1] = *(const v8h*)(arow + kb + lhi * 16 + 8);
        b.h[0] = *(const v8h*)(brow + kb + lhi * 16);
        b.h[1] = *(const v8h*)(brow + kb + lhi * 16 + 8);
        c = __builtin_amdgcn_wmma_f32_16x16x32_f16(false, a.v, false, b.v,
                                                   (short)0, c, false, false);
    }
    // Phase 2: s_in @ W_rec, K = 2048 (B columns offset by D inside Wt rows)
    for (int kb = 0; kb < LSM_R; kb += 32) {
        H16 a, b;
        a.h[0] = *(const v8h*)(srow + kb + lhi * 16);
        a.h[1] = *(const v8h*)(srow + kb + lhi * 16 + 8);
        b.h[0] = *(const v8h*)(brow + LSM_D + kb + lhi * 16);
        b.h[1] = *(const v8h*)(brow + LSM_D + kb + lhi * 16 + 8);
        c = __builtin_amdgcn_wmma_f32_16x16x32_f16(false, a.v, false, b.v,
                                                   (short)0, c, false, false);
    }

    // LIF update. C/D layout: VGPR e, lanes 0-15 -> (M=e, N=lane),
    // lanes 16-31 -> (M=e+8, N=lane-16). This wave exclusively owns its tile.
    const int n = n0 + lrow;
    const int mb = m0 + lhi * 8;
#pragma unroll
    for (int e = 0; e < 8; ++e) {
        const size_t idx = (size_t)(mb + e) * LSM_R + n;
        float un = ALPHA_F * u[idx] + c[e];
        float sp = (un >= 1.0f) ? 1.0f : 0.0f;
        u[idx]   = un * (1.0f - sp);
        cnt[idx] += sp;
        Sout[idx] = (_Float16)sp;
    }
}

__global__ __launch_bounds__(256) void lsm_finalize(const float* __restrict__ cnt,
                                                    float* __restrict__ out, int n) {
    int i = blockIdx.x * 256 + threadIdx.x;
    if (i < n) out[i] = cnt[i] * (1.0f / (float)LSM_T);
}

// ---------------------------------------------------------------------------
extern "C" void kernel_launch(void* const* d_in, const int* in_sizes, int n_in,
                              void* d_out, int out_size, void* d_ws, size_t ws_size,
                              hipStream_t stream) {
    const float* x     = (const float*)d_in[0];  // [64,256,512]
    const float* W_in  = (const float*)d_in[1];  // [512,2048]
    const float* W_rec = (const float*)d_in[2];  // [2048,2048]
    float* out = (float*)d_out;                  // [64,2048]

    char* ws = (char*)d_ws;
    // Workspace layout (all 16B-aligned, total ~28.8 MB)
    _Float16* Xh = (_Float16*)(ws + 0);            // 64*256*512 f16 = 16 MB
    _Float16* Wt = (_Float16*)(ws + 16777216);     // 2048*2560 f16  = 10 MB
    _Float16* S0 = (_Float16*)(ws + 27262976);     // 64*2048 f16 = 256 KB
    _Float16* S1 = (_Float16*)(ws + 27525120);     // 64*2048 f16 = 256 KB
    float*    Uu = (float*)   (ws + 27787264);     // 64*2048 f32 = 512 KB
    float*    Cn = (float*)   (ws + 28311552);     // 64*2048 f32 = 512 KB

    const int nX  = LSM_B * LSM_T * LSM_D;  // 8388608
    const int nW  = LSM_R * LSM_K;          // 5242880
    const int nBR = LSM_B * LSM_R;          // 131072

    lsm_convert_x<<<(nX + 255) / 256, 256, 0, stream>>>(x, Xh, nX);
    lsm_build_wt<<<(nW + 255) / 256, 256, 0, stream>>>(W_in, W_rec, Wt);
    lsm_init<<<(nBR + 255) / 256, 256, 0, stream>>>(Uu, Cn, S0, S1, nBR);

    for (int t = 0; t < LSM_T; ++t) {
        _Float16* sin  = (t & 1) ? S1 : S0;
        _Float16* sout = (t & 1) ? S0 : S1;
        lsm_step<<<64, 256, 0, stream>>>(Xh, Wt, sin, sout, Uu, Cn, t);
    }

    lsm_finalize<<<(nBR + 255) / 256, 256, 0, stream>>>(Cn, out, nBR);
}
